// GearNetEdge_38311108280750
// MI455X (gfx1250) — compile-verified, matrix operand order checked by reading, not run on hardware
//
#include <hip/hip_runtime.h>
#include <stdint.h>

// ---------------- problem constants (match reference) ----------------
#define N_NODES   100000
#define N_EDGE    500000
#define N_REDGE   1000000
#define DIM       256
#define LAYERS    3
#define R_NODE    7
#define R_EDGE    8
#define KDIM      2048      // R_EDGE*DIM == (R_NODE+1)*DIM
#define BN_EPS    1e-5f

typedef __attribute__((ext_vector_type(16))) __bf16 v16bf;
typedef __attribute__((ext_vector_type(8)))  float  v8f;

// ---------------- small helpers ----------------
__device__ __forceinline__ void atom_pk_bf16(__bf16* p, uint32_t v) {
  unsigned long long a = (unsigned long long)(uintptr_t)p;
  asm volatile("global_atomic_pk_add_bf16 %0, %1, off" :: "v"(a), "v"(v) : "memory");
}

__device__ __forceinline__ uint32_t pack_bf16(float lo, float hi) {
  union { __bf16 b[2]; uint32_t u; } u;
  u.b[0] = (__bf16)lo; u.b[1] = (__bf16)hi;
  return u.u;
}

// async 16B global -> LDS copy per lane (ASYNCcnt-tracked)
__device__ __forceinline__ void async_b128(uint32_t lds_off, const void* g) {
  unsigned long long a = (unsigned long long)(uintptr_t)g;
  asm volatile("global_load_async_to_lds_b128 %0, %1, off" :: "v"(lds_off), "v"(a) : "memory");
}
__device__ __forceinline__ void wait_async0() {
  asm volatile("s_wait_asynccnt 0x0" ::: "memory");
}

// ---------------- utility kernels ----------------
__global__ void zero_kernel(uint4* p, long n4) {
  long i = (long)blockIdx.x * blockDim.x + threadIdx.x;
  long stride = (long)gridDim.x * blockDim.x;
  const uint4 z = {0u,0u,0u,0u};
  for (; i < n4; i += stride) p[i] = z;
}

__global__ void copy_f32_kernel(const uint4* __restrict__ s, uint4* __restrict__ d, long n4) {
  long i = (long)blockIdx.x * blockDim.x + threadIdx.x;
  long stride = (long)gridDim.x * blockDim.x;
  for (; i < n4; i += stride) d[i] = s[i];
}

__global__ void cvt_f32_bf16_kernel(const float* __restrict__ s, __bf16* __restrict__ d, long n) {
  long i = ((long)blockIdx.x * blockDim.x + threadIdx.x) * 4;
  long stride = (long)gridDim.x * blockDim.x * 4;
  for (; i < n; i += stride) {
    float4 v = *(const float4*)(s + i);
    d[i+0] = (__bf16)v.x; d[i+1] = (__bf16)v.y;
    d[i+2] = (__bf16)v.z; d[i+3] = (__bf16)v.w;
  }
}

// ---------------- line-graph scatter: aggE[rdst, rtype*D + k] += mjir[rsrc, k]
__global__ void scatter_edge_kernel(const __bf16* __restrict__ mjir,
                                    const int* __restrict__ rsrc,
                                    const int* __restrict__ rdst,
                                    const int* __restrict__ rtype,
                                    __bf16* __restrict__ aggE) {
  int gid  = blockIdx.x * blockDim.x + threadIdx.x;
  int e    = gid >> 5;
  int lane = gid & 31;
  if (e >= N_REDGE) return;
  int s = rsrc[e], d = rdst[e], t = rtype[e];
  const uint4 v = *(const uint4*)(mjir + (size_t)s * DIM + lane * 8);   // 8 bf16
  __bf16* dst = aggE + (size_t)d * KDIM + t * DIM + lane * 8;
  atom_pk_bf16(dst + 0, v.x);
  atom_pk_bf16(dst + 2, v.y);
  atom_pk_bf16(dst + 4, v.z);
  atom_pk_bf16(dst + 6, v.w);
}

// ---------------- node scatter: aggN[dst, etype*D+k] += h[src,k]; aggN[dst, R*D+k] += mjir[e,k]
__global__ void scatter_node_kernel(const float* __restrict__ h,
                                    const __bf16* __restrict__ mjir,
                                    const int* __restrict__ src,
                                    const int* __restrict__ dst_,
                                    const int* __restrict__ etype,
                                    __bf16* __restrict__ aggN) {
  int gid  = blockIdx.x * blockDim.x + threadIdx.x;
  int e    = gid >> 5;
  int lane = gid & 31;
  if (e >= N_EDGE) return;
  int s = src[e], d = dst_[e], t = etype[e];
  const float4 a0 = *(const float4*)(h + (size_t)s * DIM + lane * 8);
  const float4 a1 = *(const float4*)(h + (size_t)s * DIM + lane * 8 + 4);
  __bf16* dn = aggN + (size_t)d * KDIM + t * DIM + lane * 8;
  atom_pk_bf16(dn + 0, pack_bf16(a0.x, a0.y));
  atom_pk_bf16(dn + 2, pack_bf16(a0.z, a0.w));
  atom_pk_bf16(dn + 4, pack_bf16(a1.x, a1.y));
  atom_pk_bf16(dn + 6, pack_bf16(a1.z, a1.w));
  const uint4 m = *(const uint4*)(mjir + (size_t)e * DIM + lane * 8);
  __bf16* dm = aggN + (size_t)d * KDIM + R_NODE * DIM + lane * 8;
  atom_pk_bf16(dm + 0, m.x);
  atom_pk_bf16(dm + 2, m.y);
  atom_pk_bf16(dm + 4, m.z);
  atom_pk_bf16(dm + 6, m.w);
}

// ---------------- weight swizzle into WMMA-B fragment order ----------------
// out[((ks*16 + nt)*32 + L)*16 + j] = bf16( W[K][n] ),
//   K = ks*32 + (L>>4)*8 + (j<8 ? j : 8+j), n = nt*16 + (L&15)
// W rows K<splitK come from Wa, else from Wb (K-splitK)
__global__ void swizzle_weights_kernel(const float* __restrict__ Wa,
                                       const float* __restrict__ Wb,
                                       int splitK, __bf16* __restrict__ out) {
  long i = (long)blockIdx.x * blockDim.x + threadIdx.x;
  if (i >= (long)KDIM * DIM) return;
  int j  = (int)(i & 15);
  int L  = (int)((i >> 4) & 31);
  int nt = (int)((i >> 9) & 15);
  int ks = (int)(i >> 13);
  int K  = ks * 32 + ((L >> 4) * 8) + (j < 8 ? j : 8 + j);
  int n  = nt * 16 + (L & 15);
  float v = (K < splitK) ? Wa[(size_t)K * DIM + n]
                         : Wb[(size_t)(K - splitK) * DIM + n];
  out[i] = (__bf16)v;
}

// ---------------- WMMA GEMM: C(M x 256) = A(M x 2048, bf16 row-major) @ Bw(swizzled)
// epilogue: relu_bf16 ? outB = bf16(relu(acc+bias)) : outF = acc+bias
__global__ __launch_bounds__(256)
void gemm_bf16_kernel(const __bf16* __restrict__ A,
                      const __bf16* __restrict__ Bw,
                      const float* __restrict__ bias,
                      float* __restrict__ outF,
                      __bf16* __restrict__ outB,
                      int M, int relu_bf16) {
  __shared__ __bf16 As[128][40];   // 32 K + pad (16B-aligned rows, conflict-free)
  __shared__ __bf16 Bs[8192];      // [nt][lane][16] fragment-ready, 16KB

  const int tid  = threadIdx.x;
  const int wave = tid >> 5;
  const int lane = tid & 31;
  const int m0   = blockIdx.x * 128;

  v8f acc[16];
  const v8f vz = {0.f,0.f,0.f,0.f,0.f,0.f,0.f,0.f};
#pragma unroll
  for (int i = 0; i < 16; ++i) acc[i] = vz;

  const int  arow   = tid >> 1;          // 0..127
  const int  ahalf  = tid & 1;           // which 16 of the 32 K values
  const bool avalid = (m0 + arow) < M;
  const __bf16* aptr = A + (size_t)(m0 + arow) * KDIM + ahalf * 16;

  // async B staging: thread t copies 64B of the contiguous 16KB K-slab
  const uint32_t bs_lds = (uint32_t)(uintptr_t)&Bs[0] + (uint32_t)tid * 64;
  const char*    bgl    = (const char*)Bw + (size_t)tid * 64;

  const int ar = wave * 16 + (lane & 15);   // wave's fragment row in As
  const int kb = (lane >> 4) * 8;           // K sub-base per ISA A layout
  const __bf16* bfr = &Bs[lane * 16];       // fragment base for this lane

  for (int ks = 0; ks < KDIM / 32; ++ks) {
    // B tile: async global->LDS (ASYNCcnt), no VGPR round-trip
    async_b128(bs_lds +  0, bgl +  0);
    async_b128(bs_lds + 16, bgl + 16);
    async_b128(bs_lds + 32, bgl + 32);
    async_b128(bs_lds + 48, bgl + 48);

    // A tile: VGPR-staged (zero-fill for out-of-range rows)
    uint4 a0 = {0u,0u,0u,0u}, a1 = {0u,0u,0u,0u};
    if (avalid) {
      a0 = *(const uint4*)(aptr);
      a1 = *(const uint4*)(aptr + 8);
      __builtin_prefetch(aptr + 32, 0, 1);   // global_prefetch_b8 next K slab
    }
    *(uint4*)&As[arow][ahalf * 16]     = a0;
    *(uint4*)&As[arow][ahalf * 16 + 8] = a1;

    wait_async0();
    __syncthreads();

    union AF { v16bf v; uint4 u[2]; };
    AF af;
    af.u[0] = *(const uint4*)&As[ar][kb];
    af.u[1] = *(const uint4*)&As[ar][kb + 16];

    // double-buffered B fragments: hide DS latency behind the matrix pipe
    AF bcur, bnxt;
    bcur.u[0] = *(const uint4*)(bfr);
    bcur.u[1] = *(const uint4*)(bfr + 8);
#pragma unroll
    for (int nt = 0; nt < 16; ++nt) {
      if (nt < 15) {
        bnxt.u[0] = *(const uint4*)(bfr + (nt + 1) * 512);
        bnxt.u[1] = *(const uint4*)(bfr + (nt + 1) * 512 + 8);
      }
      acc[nt] = __builtin_amdgcn_wmma_f32_16x16x32_bf16(
          false, af.v, false, bcur.v, (short)0, acc[nt], false, false);
      bcur = bnxt;
    }
    __syncthreads();

    aptr += 32;
    bgl  += 16384;
  }

  // epilogue: C layout — VGPR v holds M = v + 8*(lane>=16), N = nt*16 + (lane&15)
  const int mbase = m0 + wave * 16 + 8 * (lane >> 4);
  const int nlo   = lane & 15;
#pragma unroll
  for (int nt = 0; nt < 16; ++nt) {
    const int ncol = nt * 16 + nlo;
    const float bv = bias[ncol];
#pragma unroll
    for (int v = 0; v < 8; ++v) {
      const int row = mbase + v;
      if (row < M) {
        float val = acc[nt][v] + bv;
        if (relu_bf16) {
          val = fmaxf(val, 0.f);
          outB[(size_t)row * DIM + ncol] = (__bf16)val;
        } else {
          outF[(size_t)row * DIM + ncol] = val;
        }
      }
    }
  }
}

// ---------------- batch-norm statistics + apply ----------------
__global__ void bn_stats_kernel(const float* __restrict__ nm, float* __restrict__ sums) {
  const int col = threadIdx.x;       // 256 threads = 256 columns
  float s = 0.f, q = 0.f;
  for (int r = blockIdx.x; r < N_NODES; r += gridDim.x) {
    float v = nm[(size_t)r * DIM + col];
    s += v; q += v * v;
  }
  atomicAdd(&sums[col], s);
  atomicAdd(&sums[DIM + col], q);
}

__global__ void bn_apply_kernel(const float* __restrict__ nm,
                                const float* __restrict__ sums,
                                const float* __restrict__ gamma,
                                const float* __restrict__ beta,
                                float* __restrict__ h) {
  long i = (long)blockIdx.x * blockDim.x + threadIdx.x;
  long stride = (long)gridDim.x * blockDim.x;
  const float inv_n = 1.0f / (float)N_NODES;
  const long total = (long)N_NODES * DIM;
  for (; i < total; i += stride) {
    int col = (int)(i & (DIM - 1));
    float mu  = sums[col] * inv_n;
    float var = sums[DIM + col] * inv_n - mu * mu;
    float rs  = rsqrtf(var + BN_EPS);
    float v   = (nm[i] - mu) * rs * gamma[col] + beta[col];
    h[i] = fmaxf(v, 0.f) + h[i];
  }
}

// ---------------- workspace layout (bytes) ----------------
static const size_t SZ_AGGE = (size_t)N_EDGE * KDIM * 2;   // 2,048,000,000
static const size_t SZ_AGGN = (size_t)N_NODES * KDIM * 2;  //   409,600,000
static const size_t SZ_MJ   = (size_t)N_EDGE * DIM * 2;    //   256,000,000
static const size_t SZ_NM   = (size_t)N_NODES * DIM * 4;   //   102,400,000
static const size_t SZ_WC   = (size_t)KDIM * DIM * 2;      //     1,048,576

static const size_t OFF_AGGE = 0;
static const size_t OFF_AGGN = OFF_AGGE + SZ_AGGE;
static const size_t OFF_MJA  = OFF_AGGN + SZ_AGGN;
static const size_t OFF_MJB  = OFF_MJA  + SZ_MJ;
static const size_t OFF_NM   = OFF_MJB  + SZ_MJ;
static const size_t OFF_WC   = OFF_NM   + SZ_NM;
static const size_t OFF_SUMS = OFF_WC   + SZ_WC;           // 512 floats

extern "C" void kernel_launch(void* const* d_in, const int* in_sizes, int n_in,
                              void* d_out, int out_size, void* d_ws, size_t ws_size,
                              hipStream_t stream) {
  (void)in_sizes; (void)n_in; (void)out_size; (void)ws_size;

  const float* x          = (const float*)d_in[0];
  const int*   edge_index = (const int*)  d_in[1];
  const int*   edge_type  = (const int*)  d_in[2];
  const float* edge_feat  = (const float*)d_in[3];
  const int*   redge_index= (const int*)  d_in[4];
  const int*   redge_type = (const int*)  d_in[5];
  const float* W_node     = (const float*)d_in[6];
  const float* b_node     = (const float*)d_in[7];
  const float* W_msg      = (const float*)d_in[8];
  const float* gamma      = (const float*)d_in[9];
  const float* beta       = (const float*)d_in[10];
  const float* W_edge     = (const float*)d_in[11];
  const float* b_edge     = (const float*)d_in[12];

  float* h = (float*)d_out;                 // h lives in d_out (init = x)
  char*  ws = (char*)d_ws;
  __bf16* aggE  = (__bf16*)(ws + OFF_AGGE);
  __bf16* aggN  = (__bf16*)(ws + OFF_AGGN);
  __bf16* mjA   = (__bf16*)(ws + OFF_MJA);
  __bf16* mjB   = (__bf16*)(ws + OFF_MJB);
  float*  nm    = (float*) (ws + OFF_NM);
  __bf16* Wc    = (__bf16*)(ws + OFF_WC);
  float*  sums  = (float*) (ws + OFF_SUMS);

  const int* rsrc = redge_index;
  const int* rdst = redge_index + N_REDGE;
  const int* esrc = edge_index;
  const int* edst = edge_index + N_EDGE;

  // h = x
  {
    long n4 = (long)N_NODES * DIM / 4;
    copy_f32_kernel<<<4096, 256, 0, stream>>>((const uint4*)x, (uint4*)h, n4);
  }
  // mjir(layer 0) = bf16(edge_feat)
  cvt_f32_bf16_kernel<<<8192, 256, 0, stream>>>(edge_feat, mjA, (long)N_EDGE * DIM);

  for (int l = 0; l < LAYERS; ++l) {
    __bf16* mj_cur  = (l & 1) ? mjB : mjA;
    __bf16* mj_next = (l & 1) ? mjA : mjB;

    // --- edge message passing: aggE <- scatter, then GEMM ---
    zero_kernel<<<16384, 256, 0, stream>>>((uint4*)aggE, (long)(SZ_AGGE / 16));
    scatter_edge_kernel<<<N_REDGE * 32 / 256, 256, 0, stream>>>(
        mj_cur, rsrc, rdst, redge_type, aggE);
    swizzle_weights_kernel<<<(KDIM * DIM) / 256, 256, 0, stream>>>(
        W_edge + (size_t)l * R_EDGE * DIM * DIM,
        W_edge + (size_t)l * R_EDGE * DIM * DIM, KDIM, Wc);
    gemm_bf16_kernel<<<(N_EDGE + 127) / 128, 256, 0, stream>>>(
        aggE, Wc, b_edge + (size_t)l * DIM, nullptr, mj_next, N_EDGE, 1);

    // --- node conv: aggN <- scatter(h, mj_next), then GEMM ---
    zero_kernel<<<16384, 256, 0, stream>>>((uint4*)aggN, (long)(SZ_AGGN / 16));
    scatter_node_kernel<<<N_EDGE * 32 / 256, 256, 0, stream>>>(
        h, mj_next, esrc, edst, edge_type, aggN);
    swizzle_weights_kernel<<<(KDIM * DIM) / 256, 256, 0, stream>>>(
        W_node + (size_t)l * R_NODE * DIM * DIM,
        W_msg + (size_t)l * DIM * DIM, R_NODE * DIM, Wc);
    gemm_bf16_kernel<<<(N_NODES + 127) / 128, 256, 0, stream>>>(
        aggN, Wc, b_node + (size_t)l * DIM, nm, nullptr, N_NODES, 0);

    // --- batch norm + relu + residual ---
    zero_kernel<<<1, 256, 0, stream>>>((uint4*)sums, (long)(2 * DIM * 4 / 16));
    bn_stats_kernel<<<512, 256, 0, stream>>>(nm, sums);
    bn_apply_kernel<<<8192, 256, 0, stream>>>(nm, sums,
        gamma + (size_t)l * DIM, beta + (size_t)l * DIM, h);
  }
}
